// SelfAttention_75222057222809
// MI455X (gfx1250) — matrix operand order repaired and checked by reading, not compiled, not run on hardware
//
#include <hip/hip_runtime.h>
#include <stdint.h>

typedef __bf16 bf16_t;
typedef __attribute__((ext_vector_type(16))) __bf16 v16bf;
typedef __attribute__((ext_vector_type(8)))  __bf16 v8bf;
typedef __attribute__((ext_vector_type(8)))  float  v8f;

#define B_ 4
#define T_ 2048
#define C_ 1024
#define H_ 16
#define D_ 64

__device__ __forceinline__ bf16_t f2bf(float f) {
  union { float f; unsigned u; } cv; cv.f = f;
  unsigned u = cv.u;
  unsigned r = u + 0x7FFFu + ((u >> 16) & 1u);   // round-to-nearest-even
  unsigned short h = (unsigned short)(r >> 16);
  union { unsigned short s; bf16_t b; } cb; cb.s = h;
  return cb.b;
}

__device__ __forceinline__ v8f wmma_bf16(v16bf a, v16bf b, v8f c) {
  // (neg_a, A, neg_b, B, c_mod, C, reuse_a, reuse_b)
  return __builtin_amdgcn_wmma_f32_16x16x32_bf16(false, a, false, b, (short)0, c,
                                                 false, false);
}

// A operand: 16(M) x 32(K) bf16 tile from row-major matrix, at (row0, k0).
// A layout: lane m=L%16, kh=L/16; VGPR0-3 => K=kh*16+0..7,
// VGPR4-7 => K=kh*16+8..15.  Two contiguous 16B chunks per lane.
__device__ __forceinline__ v16bf load_a(const bf16_t* p, int lda, int row0, int k0) {
  int lane = threadIdx.x & 31;
  int m = lane & 15, kh = lane >> 4;
  const bf16_t* base = p + (size_t)(row0 + m) * lda + k0 + kh * 16;
  v8bf lo = *(const v8bf*)(base);
  v8bf hi = *(const v8bf*)(base + 8);
  v16bf a;
#pragma unroll
  for (int e = 0; e < 8; ++e) { a[e] = lo[e]; a[8 + e] = hi[e]; }
  return a;
}

// B operand: 32(K) x 16(N) with B[k,n] = W[col0+n, k0+k] (W row-major).
// 16-bit B layout used here (contiguous k, same as load_a): lane n=L%16, kh=L/16; VGPR j => K=kh*16+2j,2j+1
// => one 32B contiguous load per lane from row (col0+n) of W.
__device__ __forceinline__ v16bf load_b(const bf16_t* w, int ldw, int col0, int k0) {
  int lane = threadIdx.x & 31;
  int n = lane & 15, kh = lane >> 4;
  return *(const v16bf*)(w + (size_t)(col0 + n) * ldw + k0 + kh * 16);
}

__global__ void cast_kernel(const float* __restrict__ src, bf16_t* __restrict__ dst, int n) {
  int i = blockIdx.x * blockDim.x + threadIdx.x;
  int stride = gridDim.x * blockDim.x;
  for (; i < n; i += stride) dst[i] = f2bf(src[i]);
}

// qkv = x @ w_attn^T ; scatter to q/k/v in [B,H,T,D] bf16, fold 1/sqrt(D) into q.
// Register-blocked: one wave computes a 32(M) x 64(N) block = 2x4 WMMA tiles,
// reusing each A operand 4x and each B operand 2x.
__global__ void qkv_kernel(const bf16_t* __restrict__ xb,
                           const bf16_t* __restrict__ wab,
                           bf16_t* __restrict__ qb,
                           bf16_t* __restrict__ kbuf,
                           bf16_t* __restrict__ vbuf) {
  const int MB = 32, NB = 64;
  const int NBT = (3 * C_) / NB;   // 48
  const int MBT = (B_ * T_) / MB;  // 256
  int wave = (int)((blockIdx.x * blockDim.x + threadIdx.x) >> 5);
  if (wave >= MBT * NBT) return;
  int bm = wave / NBT, bn = wave % NBT;
  int m0 = bm * MB, n0 = bn * NB;
  int lane = threadIdx.x & 31;
  int col = lane & 15, hi = lane >> 4;

  v8f acc[2][4] = {};
  for (int k0 = 0; k0 < C_; k0 += 32) {
    if (k0 + 64 < C_) {  // gfx1250 global_prefetch_b8 for the K-step after next
      __builtin_prefetch(xb + (size_t)(m0 + col) * C_ + k0 + 64, 0, 1);
      __builtin_prefetch(wab + (size_t)(n0 + lane) * C_ + k0 + 64, 0, 1);
    }
    v16bf a0 = load_a(xb, C_, m0,      k0);
    v16bf a1 = load_a(xb, C_, m0 + 16, k0);
    v16bf b0 = load_b(wab, C_, n0,      k0);
    v16bf b1 = load_b(wab, C_, n0 + 16, k0);
    v16bf b2 = load_b(wab, C_, n0 + 32, k0);
    v16bf b3 = load_b(wab, C_, n0 + 48, k0);
    acc[0][0] = wmma_bf16(a0, b0, acc[0][0]);
    acc[0][1] = wmma_bf16(a0, b1, acc[0][1]);
    acc[0][2] = wmma_bf16(a0, b2, acc[0][2]);
    acc[0][3] = wmma_bf16(a0, b3, acc[0][3]);
    acc[1][0] = wmma_bf16(a1, b0, acc[1][0]);
    acc[1][1] = wmma_bf16(a1, b1, acc[1][1]);
    acc[1][2] = wmma_bf16(a1, b2, acc[1][2]);
    acc[1][3] = wmma_bf16(a1, b3, acc[1][3]);
  }

#pragma unroll
  for (int i = 0; i < 2; ++i) {
#pragma unroll
    for (int j = 0; j < 4; ++j) {
      int f = n0 + j * 16 + col;
      int s = f >> 10;                 // 0=q 1=k 2=v
      int cc = f & (C_ - 1);
      int h = cc >> 6, d = cc & 63;
      bf16_t* dst = (s == 0) ? qb : (s == 1) ? kbuf : vbuf;
      float scl = (s == 0) ? 0.125f : 1.0f;   // 1/sqrt(64)
#pragma unroll
      for (int v = 0; v < 8; ++v) {
        int gm = m0 + i * 16 + v + 8 * hi;   // token row in [0, B*T)
        int bi = gm >> 11;                   // / T_
        int t = gm & (T_ - 1);
        dst[(((size_t)(bi * H_ + h)) * T_ + t) * D_ + d] = f2bf(acc[i][j][v] * scl);
      }
    }
  }
}

// Flash attention: one wave handles 16 queries of one (b,h).
// Scores computed transposed (S^T = K * Q^T) so per-query softmax reduction is
// 8 in-lane values + one shfl_xor(16); output accumulated transposed
// (O^T += V^T * P^T) in 4 d-tiles.
__global__ void attn_kernel(const bf16_t* __restrict__ qb,
                            const bf16_t* __restrict__ kbuf,
                            const bf16_t* __restrict__ vbuf,
                            bf16_t* __restrict__ att) {
  const int QT = T_ / 16;                     // 128
  int wave = (int)((blockIdx.x * blockDim.x + threadIdx.x) >> 5);
  if (wave >= B_ * H_ * QT) return;
  int qt = wave & (QT - 1);
  int bh = wave >> 7;
  int qbase = qt * 16;
  const bf16_t* qp = qb   + (size_t)bh * T_ * D_;
  const bf16_t* kp = kbuf + (size_t)bh * T_ * D_;
  const bf16_t* vp = vbuf + (size_t)bh * T_ * D_;
  int lane = threadIdx.x & 31;
  int qcol = lane & 15, hi = lane >> 4;
  int qi = qbase + qcol;
  const float NEG_INF = -__builtin_inff();

  v16bf bq0 = load_b(qp, D_, qbase, 0);       // Q^T, d = 0..31
  v16bf bq1 = load_b(qp, D_, qbase, 32);      // Q^T, d = 32..63

  float m_run = NEG_INF, l_run = 0.0f;
  v8f o[4] = {};

  for (int kb = 0; kb < qbase + 16; kb += 32) {
    // S^T tiles: keys [kb,kb+16) and [kb+16,kb+32) x 16 queries
    v8f s0 = {}, s1 = {};
    {
      v16bf a = load_a(kp, D_, kb, 0);        s0 = wmma_bf16(a, bq0, s0);
      a = load_a(kp, D_, kb, 32);             s0 = wmma_bf16(a, bq1, s0);
      a = load_a(kp, D_, kb + 16, 0);         s1 = wmma_bf16(a, bq0, s1);
      a = load_a(kp, D_, kb + 16, 32);        s1 = wmma_bf16(a, bq1, s1);
    }
    // causal mask + running max (per query column)
    float p0[8], p1[8];
    float mx = NEG_INF;
#pragma unroll
    for (int v = 0; v < 8; ++v) {
      int k0i = kb + v + 8 * hi;
      int k1i = k0i + 16;
      float x0 = (k0i <= qi) ? s0[v] : NEG_INF;
      float x1 = (k1i <= qi) ? s1[v] : NEG_INF;
      p0[v] = x0; p1[v] = x1;
      mx = fmaxf(mx, fmaxf(x0, x1));
    }
    mx = fmaxf(mx, __shfl_xor(mx, 16, 32));
    float m_new = fmaxf(m_run, mx);           // finite from first chunk on
    float scale = __expf(m_run - m_new);
    float ls = 0.0f;
#pragma unroll
    for (int v = 0; v < 8; ++v) {
      p0[v] = __expf(p0[v] - m_new);
      p1[v] = __expf(p1[v] - m_new);
      ls += p0[v] + p1[v];
    }
    ls += __shfl_xor(ls, 16, 32);
    l_run = l_run * scale + ls;
    m_run = m_new;
#pragma unroll
    for (int t = 0; t < 4; ++t)
#pragma unroll
      for (int v = 0; v < 8; ++v) o[t][v] *= scale;

    // Pack P^T (32 keys x 16 queries) into a B operand: lane L (col q=L%16)
    // needs rows 0..15 of tile[hi]; its pair lane (L^16) holds the other 8 rows.
    v16bf bp;
#pragma unroll
    for (int v = 0; v < 8; ++v) {
      float send = hi ? p0[v] : p1[v];
      float recv = __shfl_xor(send, 16, 32);
      float lo8 = hi ? recv : p0[v];          // rows 0..7 of tile[hi]
      float hi8 = hi ? p1[v] : recv;          // rows 8..15 of tile[hi]
      bp[v]     = f2bf(lo8);
      bp[8 + v] = f2bf(hi8);
    }

    // O^T += V^T(16d x 32k) * P^T(32k x 16q), 4 d-chunks
#pragma unroll
    for (int td = 0; td < 4; ++td) {
      int d = td * 16 + qcol;                 // A-row m = lane%16
      v16bf a;
#pragma unroll
      for (int e = 0; e < 8; ++e) {
        a[e]     = vp[(size_t)(kb + hi * 16 + e) * D_ + d];
        a[8 + e] = vp[(size_t)(kb + hi * 16 + 8 + e) * D_ + d];
      }
      o[td] = wmma_bf16(a, bp, o[td]);
    }
  }

  float inv = 1.0f / l_run;
  int b = bh >> 4, h = bh & 15;
  int t = qbase + qcol;
  bf16_t* dst = att + ((size_t)(b * T_ + t)) * C_ + h * D_;
#pragma unroll
  for (int td = 0; td < 4; ++td)
#pragma unroll
    for (int v = 0; v < 8; ++v) {
      int d = td * 16 + v + 8 * hi;
      dst[d] = f2bf(o[td][v] * inv);
    }
}

// out = att @ w_proj^T, f32 output.  Same 32x64 register blocking.
__global__ void proj_kernel(const bf16_t* __restrict__ att,
                            const bf16_t* __restrict__ wpb,
                            float* __restrict__ out) {
  const int MB = 32, NB = 64;
  const int NBT = C_ / NB;         // 16
  const int MBT = (B_ * T_) / MB;  // 256
  int wave = (int)((blockIdx.x * blockDim.x + threadIdx.x) >> 5);
  if (wave >= MBT * NBT) return;
  int bm = wave / NBT, bn = wave % NBT;
  int m0 = bm * MB, n0 = bn * NB;
  int lane = threadIdx.x & 31;
  int col = lane & 15, hi = lane >> 4;

  v8f acc[2][4] = {};
  for (int k0 = 0; k0 < C_; k0 += 32) {
    if (k0 + 64 < C_) {
      __builtin_prefetch(att + (size_t)(m0 + col) * C_ + k0 + 64, 0, 1);
      __builtin_prefetch(wpb + (size_t)(n0 + lane) * C_ + k0 + 64, 0, 1);
    }
    v16bf a0 = load_a(att, C_, m0,      k0);
    v16bf a1 = load_a(att, C_, m0 + 16, k0);
    v16bf b0 = load_b(wpb, C_, n0,      k0);
    v16bf b1 = load_b(wpb, C_, n0 + 16, k0);
    v16bf b2 = load_b(wpb, C_, n0 + 32, k0);
    v16bf b3 = load_b(wpb, C_, n0 + 48, k0);
    acc[0][0] = wmma_bf16(a0, b0, acc[0][0]);
    acc[0][1] = wmma_bf16(a0, b1, acc[0][1]);
    acc[0][2] = wmma_bf16(a0, b2, acc[0][2]);
    acc[0][3] = wmma_bf16(a0, b3, acc[0][3]);
    acc[1][0] = wmma_bf16(a1, b0, acc[1][0]);
    acc[1][1] = wmma_bf16(a1, b1, acc[1][1]);
    acc[1][2] = wmma_bf16(a1, b2, acc[1][2]);
    acc[1][3] = wmma_bf16(a1, b3, acc[1][3]);
  }

#pragma unroll
  for (int i = 0; i < 2; ++i)
#pragma unroll
    for (int j = 0; j < 4; ++j)
#pragma unroll
      for (int v = 0; v < 8; ++v) {
        int gm = m0 + i * 16 + v + 8 * hi;
        out[(size_t)gm * C_ + n0 + j * 16 + col] = acc[i][j][v];
      }
}

extern "C" void kernel_launch(void* const* d_in, const int* in_sizes, int n_in,
                              void* d_out, int out_size, void* d_ws, size_t ws_size,
                              hipStream_t stream) {
  const float* x      = (const float*)d_in[0];
  const float* w_attn = (const float*)d_in[1];
  const float* w_proj = (const float*)d_in[2];
  float* out = (float*)d_out;

  const size_t NX  = (size_t)B_ * T_ * C_;   // 8,388,608
  const size_t NWA = (size_t)3 * C_ * C_;    // 3,145,728
  const size_t NWP = (size_t)C_ * C_;        // 1,048,576

  bf16_t* xb   = (bf16_t*)d_ws;
  bf16_t* wab  = xb  + NX;
  bf16_t* wpb  = wab + NWA;
  bf16_t* qb   = wpb + NWP;
  bf16_t* kbuf = qb  + NX;
  bf16_t* vbuf = kbuf + NX;
  bf16_t* att  = vbuf + NX;
  (void)in_sizes; (void)n_in; (void)out_size; (void)ws_size;

  cast_kernel<<<2048, 256, 0, stream>>>(x,      xb,  (int)NX);
  cast_kernel<<<2048, 256, 0, stream>>>(w_attn, wab, (int)NWA);
  cast_kernel<<<512,  256, 0, stream>>>(w_proj, wpb, (int)NWP);

  // 256 x 48 32x64-blocks, 8 waves per 256-thread block
  qkv_kernel<<<(256 * 48) / 8, 256, 0, stream>>>(xb, wab, qb, kbuf, vbuf);
  // B*H*(T/16) = 8192 waves
  attn_kernel<<<(B_ * H_ * (T_ / 16)) / 8, 256, 0, stream>>>(qb, kbuf, vbuf, att);
  // 256 x 16 32x64-blocks
  proj_kernel<<<(256 * 16) / 8, 256, 0, stream>>>(att, wpb, out);
}
